// S_DNA_different_Attention_90185723281952
// MI455X (gfx1250) — matrix operand order repaired and checked by reading, not compiled, not run on hardware
//
#include <hip/hip_runtime.h>

typedef __attribute__((ext_vector_type(16))) __bf16 v16bf;
typedef __attribute__((ext_vector_type(8)))  __bf16 v8bf;
typedef __attribute__((ext_vector_type(8)))  float  v8f;

#define DIM    1024
#define BATCH  2048
#define SEQ    41
#define SPAD   48          // M padded to 3 WMMA tiles
#define LDSTR  1040        // LDS row stride in bf16 elements (1024 + 16 pad)

// ---------- helpers ----------
// native converts: let the backend emit v_cvt_pk_bf16_f32 (RNE)
static __device__ __forceinline__ __bf16 f2bf(float f) { return (__bf16)f; }
static __device__ __forceinline__ unsigned short f2bf_u(float f) {
    union { __bf16 b; unsigned short s; } v;
    v.b = (__bf16)f;
    return v.s;
}
static __device__ __forceinline__ unsigned pack2bf(float a, float b) {
    union { __bf16 h[2]; unsigned u; } v;
    v.h[0] = (__bf16)a; v.h[1] = (__bf16)b;
    return v.u;
}
static __device__ __forceinline__ float bfu2f(unsigned short h) {
    return __uint_as_float(((unsigned)h) << 16);
}
static __device__ __forceinline__ v8f vzero8() {
    v8f z;
#pragma unroll
    for (int i = 0; i < 8; ++i) z[i] = 0.0f;
    return z;
}
// wave32 XOR-butterfly sum via ds_swizzle (group-of-32 mode: and=0x1f, xor=m)
static __device__ __forceinline__ float wave_sum(float v) {
    v += __int_as_float(__builtin_amdgcn_ds_swizzle(__float_as_int(v), (16 << 10) | 0x1f));
    v += __int_as_float(__builtin_amdgcn_ds_swizzle(__float_as_int(v), (8  << 10) | 0x1f));
    v += __int_as_float(__builtin_amdgcn_ds_swizzle(__float_as_int(v), (4  << 10) | 0x1f));
    v += __int_as_float(__builtin_amdgcn_ds_swizzle(__float_as_int(v), (2  << 10) | 0x1f));
    v += __int_as_float(__builtin_amdgcn_ds_swizzle(__float_as_int(v), (1  << 10) | 0x1f));
    return v;
}
static __device__ __forceinline__ v16bf cat16(v8bf lo, v8bf hi) {
    return __builtin_shufflevector(lo, hi, 0,1,2,3,4,5,6,7,8,9,10,11,12,13,14,15);
}

// ---------- kernel 0: Wv fp32 -> bf16 ----------
__global__ void __launch_bounds__(256) k_cvt(const float* __restrict__ w,
                                             unsigned short* __restrict__ o, int n) {
    int i = blockIdx.x * blockDim.x + threadIdx.x;
    for (; i < n; i += gridDim.x * blockDim.x) o[i] = f2bf_u(w[i]);
}

// ---------- kernel 1: Q = query @ Wq^T + bq  (WMMA bf16) ----------
// grid (8, 32): x -> 128-col slab, y -> 64-row slab; 256 threads = 8 waves
__global__ void __launch_bounds__(256) k_qproj(const float* __restrict__ query,
                                               const float* __restrict__ wq,
                                               const float* __restrict__ bq,
                                               float* __restrict__ Qf,
                                               unsigned short* __restrict__ Qb) {
    const int w    = threadIdx.x >> 5;
    const int lane = threadIdx.x & 31;
    const int lr   = lane & 15;
    const int kh   = lane >> 4;
    const int m0   = blockIdx.y * 64;
    const int ncol = blockIdx.x * 128 + w * 16 + lr;   // B column == Wq row (contiguous)

    v8f acc[4];
#pragma unroll
    for (int mt = 0; mt < 4; ++mt) acc[mt] = vzero8();

    for (int kb = 0; kb < DIM; kb += 32) {
        v16bf bfrag;
        {
            const float* bp = wq + (size_t)ncol * DIM + kb + kh * 8;  // K = kb + kh*8 + j (j<8), kb + 16 + kh*8 + j-8 (j>=8)
#pragma unroll
            for (int j = 0; j < 16; ++j) bfrag[j] = f2bf(bp[j < 8 ? j : j + 8]);
        }
#pragma unroll
        for (int mt = 0; mt < 4; ++mt) {
            const float* ap = query + (size_t)(m0 + mt * 16 + lr) * DIM + kb + kh * 8;
            v16bf afrag;
#pragma unroll
            for (int j = 0; j < 8; ++j) {                // elems 0..7: K=kb+kh*8+j
                afrag[j]     = f2bf(ap[j]);              // elems 8..15: K=kb+16+kh*8+j
                afrag[j + 8] = f2bf(ap[16 + j]);
            }
            acc[mt] = __builtin_amdgcn_wmma_f32_16x16x32_bf16(
                false, afrag, false, bfrag, (short)0, acc[mt], false, false);
        }
    }
    const float bias = bq[ncol];
#pragma unroll
    for (int mt = 0; mt < 4; ++mt) {
#pragma unroll
        for (int r = 0; r < 8; ++r) {
            int m = m0 + mt * 16 + r + 8 * kh;           // C/D: VGPR r -> M = r + 8*kh
            float v = acc[mt][r] + bias;
            Qf[(size_t)m * DIM + ncol] = v;
            Qb[(size_t)m * DIM + ncol] = f2bf_u(v);
        }
    }
}

// ---------- kernel 2: qdot[b][s] = Q[b] . bk_s  (3 waves/block) ----------
__global__ void __launch_bounds__(96) k_qdot(const float* __restrict__ Qf,
                                             const float* __restrict__ b1,
                                             const float* __restrict__ b2,
                                             const float* __restrict__ b3,
                                             float* __restrict__ qdot) {
    const int b = blockIdx.x;
    const int s = threadIdx.x >> 5, lane = threadIdx.x & 31;
    const float* bias = (s == 0) ? b1 : ((s == 1) ? b2 : b3);
    float p = 0.0f;
    for (int d = lane; d < DIM; d += 32) p += Qf[(size_t)b * DIM + d] * bias[d];
    p = wave_sum(p);
    if (lane == 0) qdot[b * 3 + s] = p;
}

// ---------- kernel 3: P_s = Q @ Wk_s  (WMMA bf16, B columns strided) ----------
// grid (8, 32, 3)
__global__ void __launch_bounds__(256) k_pproj(const unsigned short* __restrict__ Qb,
                                               const float* __restrict__ wk1,
                                               const float* __restrict__ wk2,
                                               const float* __restrict__ wk3,
                                               float* __restrict__ P) {
    const int seg = blockIdx.z;
    const float* wk = (seg == 0) ? wk1 : ((seg == 1) ? wk2 : wk3);
    const int w    = threadIdx.x >> 5;
    const int lane = threadIdx.x & 31;
    const int lr   = lane & 15;
    const int kh   = lane >> 4;
    const int m0   = blockIdx.y * 64;
    const int ncol = blockIdx.x * 128 + w * 16 + lr;

    v8f acc[4];
#pragma unroll
    for (int mt = 0; mt < 4; ++mt) acc[mt] = vzero8();

    for (int kb = 0; kb < DIM; kb += 32) {
        v16bf bfrag;
        {
            // B[k][n] = wk[k*DIM + n] -> column n is strided by DIM
            const float* bp = wk + (size_t)(kb + kh * 8) * DIM + ncol;
#pragma unroll
            for (int j = 0; j < 16; ++j) bfrag[j] = f2bf(bp[(size_t)(j < 8 ? j : j + 8) * DIM]);
        }
#pragma unroll
        for (int mt = 0; mt < 4; ++mt) {
            const unsigned short* row = Qb + (size_t)(m0 + mt * 16 + lr) * DIM;
            v8bf lo = *(const v8bf*)(row + kb + kh * 8);
            v8bf hi = *(const v8bf*)(row + kb + 16 + kh * 8);
            v16bf afrag = cat16(lo, hi);
            acc[mt] = __builtin_amdgcn_wmma_f32_16x16x32_bf16(
                false, afrag, false, bfrag, (short)0, acc[mt], false, false);
        }
    }
    float* Ps = P + (size_t)seg * BATCH * DIM;
#pragma unroll
    for (int mt = 0; mt < 4; ++mt) {
#pragma unroll
        for (int r = 0; r < 8; ++r) {
            int m = m0 + mt * 16 + r + 8 * kh;
            Ps[(size_t)m * DIM + ncol] = acc[mt][r];
        }
    }
}

// ---------- kernel 4: fused scores + softmax + V GEMM + gate ----------
// one workgroup per batch; enc[b] staged once into LDS as bf16
__global__ void __launch_bounds__(256) k_attn(const float* __restrict__ enc,
                                              const float* __restrict__ P,
                                              const float* __restrict__ qdot,
                                              const unsigned short* __restrict__ wvb,
                                              const float* __restrict__ bv,
                                              float* __restrict__ out) {
    extern __shared__ unsigned char smem_raw[];
    unsigned short* encS = (unsigned short*)smem_raw;                  // SPAD*LDSTR bf16
    float* sS    = (float*)(smem_raw + (size_t)SPAD * LDSTR * 2);      // 64 floats
    float* probS = sS + 64;                                            // 64 floats

    const int b    = blockIdx.x;
    const int t    = threadIdx.x;
    const int w    = t >> 5;
    const int lane = t & 31;
    const int lr   = lane & 15;
    const int kh   = lane >> 4;

    // ---- phase 1: enc[b] fp32 -> LDS bf16 (rows 0..40), zero-pad 41..47 ----
    const float4* ep = (const float4*)(enc + (size_t)b * SEQ * DIM);
    for (int i4 = t; i4 < (SEQ * DIM) / 4; i4 += 256) {
        float4 v = ep[i4];
        int row = i4 >> 8;              // (i4*4)/1024
        int col = (i4 & 255) << 2;      // multiple of 4 -> 8B-aligned LDS store
        uint2 pk;
        pk.x = pack2bf(v.x, v.y);
        pk.y = pack2bf(v.z, v.w);
        *(uint2*)(&encS[row * LDSTR + col]) = pk;
    }
    {   // zero rows 41..47 with dword stores
        unsigned* zp = (unsigned*)(&encS[SEQ * LDSTR]);
        for (int i = t; i < ((SPAD - SEQ) * LDSTR) / 2; i += 256) zp[i] = 0u;
    }
    __syncthreads();

    // ---- phase 2: s[k] = enc[b,k] . P_seg(k)[b] + Q.bk  (per-wave dots) ----
    for (int k = w; k < SEQ; k += 8) {
        int seg = (k < 10) ? 0 : ((k < 31) ? 1 : 2);
        const float* Pp = P + ((size_t)seg * BATCH + b) * DIM;
        float p = 0.0f;
        for (int e = lane * 2; e < DIM; e += 64) {
            unsigned pr = *(const unsigned*)(&encS[k * LDSTR + e]);
            float2 pv = *(const float2*)(&Pp[e]);
            p += bfu2f((unsigned short)(pr & 0xFFFFu)) * pv.x;
            p += bfu2f((unsigned short)(pr >> 16))     * pv.y;
        }
        p = wave_sum(p);
        if (lane == 0) {
            float mix = (seg == 1) ? 0.6f : 0.2f;
            sS[k] = mix * (p + qdot[b * 3 + seg]) * (1.0f / 32.0f);
        }
    }
    __syncthreads();

    // ---- phase 3: softmax over 41 (tiny; single thread) ----
    if (t == 0) {
        float mx = sS[0];
        for (int k = 1; k < SEQ; ++k) mx = fmaxf(mx, sS[k]);
        float sum = 0.0f;
        for (int k = 0; k < SEQ; ++k) { float e = __expf(sS[k] - mx); probS[k] = e; sum += e; }
        float inv = 1.0f / sum;
        for (int k = 0; k < SEQ; ++k) probS[k] *= inv;
    }
    __syncthreads();

    // ---- phase 4: out[b] = probs ⊙ (enc[b] @ Wv^T + bv)  via WMMA ----
    float* ob = out + (size_t)b * SEQ * DIM;
    for (int g = 0; g < 2; ++g) {
        const int nb = w * 128 + g * 64;    // this wave's 64-column slab
        v8f acc[3][4];
#pragma unroll
        for (int mt = 0; mt < 3; ++mt)
#pragma unroll
            for (int nt = 0; nt < 4; ++nt) acc[mt][nt] = vzero8();

        // software pipeline: prefetch next k-step's B fragments during WMMAs
        const unsigned short* bbase[4];
#pragma unroll
        for (int nt = 0; nt < 4; ++nt)
            bbase[nt] = wvb + (size_t)(nb + nt * 16 + lr) * DIM + kh * 16;

        v16bf bcur[4];
#pragma unroll
        for (int nt = 0; nt < 4; ++nt) bcur[nt] = *(const v16bf*)(bbase[nt]);

        for (int kb = 0; kb < DIM; kb += 32) {
            v16bf bnext[4];
            if (kb + 32 < DIM) {
#pragma unroll
                for (int nt = 0; nt < 4; ++nt)
                    bnext[nt] = *(const v16bf*)(bbase[nt] + kb + 32);
            }
            v16bf af[3];
#pragma unroll
            for (int mt = 0; mt < 3; ++mt) {
                const unsigned short* row = &encS[(mt * 16 + lr) * LDSTR];
                v8bf lo = *(const v8bf*)(row + kb + kh * 16);
                v8bf hi = *(const v8bf*)(row + kb + kh * 16 + 8);
                af[mt] = cat16(lo, hi);
            }
#pragma unroll
            for (int nt = 0; nt < 4; ++nt)
#pragma unroll
                for (int mt = 0; mt < 3; ++mt)
                    acc[mt][nt] = __builtin_amdgcn_wmma_f32_16x16x32_bf16(
                        false, af[mt], false, bcur[nt], (short)0, acc[mt][nt], false, false);
            if (kb + 32 < DIM) {
#pragma unroll
                for (int nt = 0; nt < 4; ++nt) bcur[nt] = bnext[nt];
            }
        }
#pragma unroll
        for (int mt = 0; mt < 3; ++mt)
#pragma unroll
            for (int nt = 0; nt < 4; ++nt) {
                int ncol = nb + nt * 16 + lr;
                float bias = bv[ncol];
#pragma unroll
                for (int r = 0; r < 8; ++r) {
                    int m = mt * 16 + r + 8 * kh;
                    if (m < SEQ)
                        ob[(size_t)m * DIM + ncol] = probS[m] * (acc[mt][nt][r] + bias);
                }
            }
    }
}

// ---------- host launcher ----------
extern "C" void kernel_launch(void* const* d_in, const int* in_sizes, int n_in,
                              void* d_out, int out_size, void* d_ws, size_t ws_size,
                              hipStream_t stream) {
    (void)in_sizes; (void)n_in; (void)out_size; (void)ws_size;
    const float* enc   = (const float*)d_in[0];
    const float* qry   = (const float*)d_in[1];
    const float* wq_w  = (const float*)d_in[2];
    const float* wq_b  = (const float*)d_in[3];
    const float* wk1_w = (const float*)d_in[4];
    const float* wk1_b = (const float*)d_in[5];
    const float* wk2_w = (const float*)d_in[6];
    const float* wk2_b = (const float*)d_in[7];
    const float* wk3_w = (const float*)d_in[8];
    const float* wk3_b = (const float*)d_in[9];
    const float* wv_w  = (const float*)d_in[10];
    const float* wv_b  = (const float*)d_in[11];
    float* out = (float*)d_out;

    unsigned char* ws = (unsigned char*)d_ws;
    unsigned short* wvb = (unsigned short*)ws;                          //  2 MB
    float*          Qf  = (float*)(ws + (size_t)(2u  << 20));           //  8 MB
    unsigned short* Qb  = (unsigned short*)(ws + (size_t)(10u << 20));  //  4 MB
    float*          P   = (float*)(ws + (size_t)(14u << 20));           // 24 MB
    float*          qd  = (float*)(ws + (size_t)(38u << 20));           // 24 KB

    k_cvt  <<<dim3(1024),      dim3(256), 0, stream>>>(wv_w, wvb, DIM * DIM);
    k_qproj<<<dim3(8, 32),     dim3(256), 0, stream>>>(qry, wq_w, wq_b, Qf, Qb);
    k_qdot <<<dim3(BATCH),     dim3(96),  0, stream>>>(Qf, wk1_b, wk2_b, wk3_b, qd);
    k_pproj<<<dim3(8, 32, 3),  dim3(256), 0, stream>>>(Qb, wk1_w, wk2_w, wk3_w, P);

    size_t smem = (size_t)SPAD * LDSTR * 2 + 128 * sizeof(float);       // ~100 KB
    k_attn <<<dim3(BATCH),     dim3(256), smem, stream>>>(enc, P, qd, wvb, wv_b, out);
}
